// Attention_52604759441672
// MI455X (gfx1250) — matrix operand order repaired and checked by reading, not compiled, not run on hardware
//
#include <hip/hip_runtime.h>
#include <hip/hip_fp16.h>

#define B_     2
#define N_     2048
#define DIM_   1024
#define H_     16
#define DH_    64
#define INNER_ 1024
#define SCALE_ 0.125f   // 64^-0.5

typedef __attribute__((ext_vector_type(16))) _Float16 v16h;
typedef __attribute__((ext_vector_type(8)))  float    v8f;

__device__ inline v8f zero8() {
    v8f z = {0.f, 0.f, 0.f, 0.f, 0.f, 0.f, 0.f, 0.f};
    return z;
}

__device__ inline v8f wmma_f16(v16h a, v16h b, v8f c) {
    return __builtin_amdgcn_wmma_f32_16x16x32_f16(
        /*neg_a=*/false, a, /*neg_b=*/false, b,
        /*c_mod=*/(short)0, c, /*reuse_a=*/false, /*reuse_b=*/false);
}

// A fragment, 16x32 f16 (M x K): lane row = lane&15, g = lane>>4.
// elements 0..7  <- K = k0 + g*16 .. +7
// elements 8..15 <- K = k0 + g*16 + 8 .. +15
__device__ inline v16h load_a_frag(const _Float16* rowptr, int k0, int g) {
    v16h a;
    uint4* a4 = reinterpret_cast<uint4*>(&a);
    a4[0] = *reinterpret_cast<const uint4*>(rowptr + k0 + g * 16);
    a4[1] = *reinterpret_cast<const uint4*>(rowptr + k0 + g * 16 + 8);
    return a;
}

// B fragment, 32x16 f16 (K x N): lane col = lane&15, K = g*16 + e (e=0..15).
// ptr must point at the 16 contiguous f16 for this lane.
__device__ inline v16h load_b_frag(const _Float16* ptr) {
    v16h b;
    uint4* b4 = reinterpret_cast<uint4*>(&b);
    b4[0] = reinterpret_cast<const uint4*>(ptr)[0];
    b4[1] = reinterpret_cast<const uint4*>(ptr)[1];
    return b;
}

// ---------------- weight convert + transpose:  Wt[n*K + k] = (f16)W[k*N + n]
__global__ void wconv_t_kernel(const float* __restrict__ W, _Float16* __restrict__ Wt,
                               int K, int Nout) {
    int idx = blockIdx.x * 256 + threadIdx.x;
    if (idx >= K * Nout) return;
    int n = idx / K;
    int k = idx - n * K;
    Wt[idx] = (_Float16)W[(size_t)k * Nout + n];
}

// ---------------- LayerNorm -> f16
__global__ void ln_kernel(const float* __restrict__ x, const float* __restrict__ gamma,
                          const float* __restrict__ beta, _Float16* __restrict__ xnh) {
    __shared__ float sbuf[256], sbuf2[256];
    int row = blockIdx.x;
    const float* xr = x + (size_t)row * DIM_;
    float s = 0.f, s2 = 0.f;
    for (int i = threadIdx.x; i < DIM_; i += 256) {
        float v = xr[i];
        s += v; s2 += v * v;
    }
    sbuf[threadIdx.x] = s; sbuf2[threadIdx.x] = s2;
    __syncthreads();
    for (int off = 128; off > 0; off >>= 1) {
        if (threadIdx.x < off) {
            sbuf[threadIdx.x]  += sbuf[threadIdx.x + off];
            sbuf2[threadIdx.x] += sbuf2[threadIdx.x + off];
        }
        __syncthreads();
    }
    float mean = sbuf[0] * (1.0f / DIM_);
    float var  = sbuf2[0] * (1.0f / DIM_) - mean * mean;
    float rstd = rsqrtf(var + 1e-5f);
    _Float16* orow = xnh + (size_t)row * DIM_;
    for (int i = threadIdx.x; i < DIM_; i += 256) {
        float v = (xr[i] - mean) * rstd * gamma[i] + beta[i];
        orow[i] = (_Float16)v;
    }
}

// ---------------- WMMA GEMM, 64x64 C macro-tile per wave (16 WMMAs / k-step)
// C[M x Nout] (f32) = A[M x K] (f16, row major) * Bt[Nout x K]^T (f16)
__global__ __launch_bounds__(32)
void gemm_f16_kernel(const _Float16* __restrict__ A, const _Float16* __restrict__ Bt,
                     float* __restrict__ C, int M, int Nout, int K) {
    int m0 = blockIdx.x * 64, n0 = blockIdx.y * 64;
    int l = threadIdx.x, g = l >> 4, t = l & 15;

    const _Float16* arow[4];
    const _Float16* brow[4];
#pragma unroll
    for (int i = 0; i < 4; ++i) {
        arow[i] = A  + (size_t)(m0 + i * 16 + t) * K;
        brow[i] = Bt + (size_t)(n0 + i * 16 + t) * K;
    }

    v8f acc[4][4];
#pragma unroll
    for (int i = 0; i < 4; ++i)
#pragma unroll
        for (int j = 0; j < 4; ++j) acc[i][j] = zero8();

    for (int k0 = 0; k0 < K; k0 += 32) {
        v16h a[4], b[4];
#pragma unroll
        for (int i = 0; i < 4; ++i) a[i] = load_a_frag(arow[i], k0, g);
#pragma unroll
        for (int j = 0; j < 4; ++j) b[j] = load_b_frag(brow[j] + k0 + g * 16);
#pragma unroll
        for (int i = 0; i < 4; ++i)
#pragma unroll
            for (int j = 0; j < 4; ++j)
                acc[i][j] = wmma_f16(a[i], b[j], acc[i][j]);
    }

#pragma unroll
    for (int i = 0; i < 4; ++i) {
        float* cp = C + (size_t)(m0 + i * 16 + g * 8) * Nout + n0 + t;
#pragma unroll
        for (int j = 0; j < 4; ++j)
#pragma unroll
            for (int v = 0; v < 8; ++v)
                cp[(size_t)v * Nout + j * 16] = acc[i][j][v];
    }
}

// ---------------- RoPE + scale + head reorder (q,k -> [b,h,n,d] f16; v -> [b,h,d,n] f16)
__global__ void rope_kernel(const float* __restrict__ q, const float* __restrict__ kv,
                            _Float16* __restrict__ qh, _Float16* __restrict__ kh,
                            _Float16* __restrict__ vt) {
    int idx = blockIdx.x * 256 + threadIdx.x;
    if (idx >= B_ * N_ * INNER_) return;
    int d = idx & (DH_ - 1);
    int h = (idx >> 6) & (H_ - 1);
    int n = (idx >> 10) & (N_ - 1);
    int b = idx >> 21;

    float tq = q[(size_t)idx];
    size_t kvrow = (size_t)(b * N_ + n) * (2 * INNER_);
    float tk = kv[kvrow + h * DH_ + d];
    float tv = kv[kvrow + INNER_ + h * DH_ + d];

    int dm = d & 31;
    float ang = (float)n * __powf(10000.f, -(float)(2 * dm) * (1.f / 64.f));
    float c = __cosf(ang), s = __sinf(ang);
    float tq2, tk2;
    if (d < 32) { tq2 = -q[(size_t)idx + 32]; tk2 = -kv[kvrow + h * DH_ + d + 32]; }
    else        { tq2 =  q[(size_t)idx - 32]; tk2 =  kv[kvrow + h * DH_ + d - 32]; }

    float qo = (tq * c + tq2 * s) * SCALE_;
    float ko = tk * c + tk2 * s;

    size_t ho = ((size_t)(b * H_ + h) * N_ + n) * DH_ + d;
    qh[ho] = (_Float16)qo;
    kh[ho] = (_Float16)ko;
    vt[((size_t)(b * H_ + h) * DH_ + d) * N_ + n] = (_Float16)tv;
}

// ---------------- causal flash attention, 32-row Q tile per wave (2 row tiles)
// K/V fragments loaded once per 32-key block and reused by both row tiles.
__global__ __launch_bounds__(32)
void attn_kernel(const _Float16* __restrict__ qh, const _Float16* __restrict__ kh,
                 const _Float16* __restrict__ vt, _Float16* __restrict__ oh) {
    int i0 = blockIdx.x * 32;
    int h = blockIdx.y, b = blockIdx.z;
    int l = threadIdx.x, g = l >> 4, t = l & 15;

    const _Float16* qb = qh + ((size_t)(b * H_ + h) * N_) * DH_;
    const _Float16* kb = kh + ((size_t)(b * H_ + h) * N_) * DH_;
    const _Float16* vb = vt + ((size_t)(b * H_ + h) * DH_) * N_;

    __shared__ __align__(16) _Float16 pt[32 * 32];   // P tiles: C-layout -> A-layout staging

    v16h qa[2][2];
#pragma unroll
    for (int r = 0; r < 2; ++r) {
        const _Float16* qrow = qb + (size_t)(i0 + r * 16 + t) * DH_;
        qa[r][0] = load_a_frag(qrow, 0, g);
        qa[r][1] = load_a_frag(qrow, 32, g);
    }

    v8f o[2][4];
    float mrow[2][8], srow[2][8];
#pragma unroll
    for (int r = 0; r < 2; ++r) {
#pragma unroll
        for (int dt = 0; dt < 4; ++dt) o[r][dt] = zero8();
#pragma unroll
        for (int v = 0; v < 8; ++v) { mrow[r][v] = -1e30f; srow[r][v] = 0.f; }
    }

    for (int jb = 0; jb < i0 + 32; jb += 32) {
        // ---- K fragments for this 32-key block (shared by both row tiles)
        const _Float16* kr0 = kb + (size_t)(jb + t) * DH_;
        const _Float16* kr1 = kb + (size_t)(jb + 16 + t) * DH_;
        v16h kf[2][2];
        kf[0][0] = load_b_frag(kr0 + g * 16);
        kf[0][1] = load_b_frag(kr0 + 32 + g * 16);
        kf[1][0] = load_b_frag(kr1 + g * 16);
        kf[1][1] = load_b_frag(kr1 + 32 + g * 16);

        // ---- S = (Q*scale) K^T : 2 row tiles x 2 key tiles, K-dim 64 in 2 steps
        v8f s[2][2];
#pragma unroll
        for (int r = 0; r < 2; ++r)
#pragma unroll
            for (int j = 0; j < 2; ++j) {
                s[r][j] = zero8();
                s[r][j] = wmma_f16(qa[r][0], kf[j][0], s[r][j]);
                s[r][j] = wmma_f16(qa[r][1], kf[j][1], s[r][j]);
            }

        // ---- causal mask + online softmax per row tile; rescale O accumulators
        __syncthreads();
#pragma unroll
        for (int r = 0; r < 2; ++r) {
#pragma unroll
            for (int v = 0; v < 8; ++v) {
                int qr = i0 + r * 16 + v + 8 * g;
                float e0 = (jb + t      <= qr) ? s[r][0][v] : -3.0e38f;
                float e1 = (jb + 16 + t <= qr) ? s[r][1][v] : -3.0e38f;
                float bm = fmaxf(e0, e1);
                bm = fmaxf(bm, __shfl_xor(bm, 1, 32));
                bm = fmaxf(bm, __shfl_xor(bm, 2, 32));
                bm = fmaxf(bm, __shfl_xor(bm, 4, 32));
                bm = fmaxf(bm, __shfl_xor(bm, 8, 32));
                float mnew = fmaxf(mrow[r][v], bm);
                float alpha = __expf(mrow[r][v] - mnew);
                mrow[r][v] = mnew;
                float p0 = __expf(e0 - mnew);
                float p1 = __expf(e1 - mnew);
                float rs = p0 + p1;
                rs += __shfl_xor(rs, 1, 32);
                rs += __shfl_xor(rs, 2, 32);
                rs += __shfl_xor(rs, 4, 32);
                rs += __shfl_xor(rs, 8, 32);
                srow[r][v] = srow[r][v] * alpha + rs;
                o[r][0][v] *= alpha;
                o[r][1][v] *= alpha;
                o[r][2][v] *= alpha;
                o[r][3][v] *= alpha;
                // stage P into LDS (C layout -> 32x32 tile)
                pt[(r * 16 + v + 8 * g) * 32 + t]      = (_Float16)p0;
                pt[(r * 16 + v + 8 * g) * 32 + 16 + t] = (_Float16)p1;
            }
        }
        __syncthreads();

        // ---- V fragments for this block (shared by both row tiles)
        v16h vf[4];
#pragma unroll
        for (int dt = 0; dt < 4; ++dt)
            vf[dt] = load_b_frag(vb + (size_t)(dt * 16 + t) * N_ + jb + g * 16);

        // ---- O += P @ V
#pragma unroll
        for (int r = 0; r < 2; ++r) {
            v16h pa;
            uint4* p4 = reinterpret_cast<uint4*>(&pa);
            p4[0] = *reinterpret_cast<const uint4*>(pt + (r * 16 + t) * 32 + g * 16);
            p4[1] = *reinterpret_cast<const uint4*>(pt + (r * 16 + t) * 32 + g * 16 + 8);
#pragma unroll
            for (int dt = 0; dt < 4; ++dt)
                o[r][dt] = wmma_f16(pa, vf[dt], o[r][dt]);
        }
    }

    // ---- 1/l normalize and store to [b, n, h*64 + d] f16 for the output GEMM
#pragma unroll
    for (int r = 0; r < 2; ++r)
#pragma unroll
        for (int v = 0; v < 8; ++v) {
            float inv = 1.0f / srow[r][v];
            size_t rowoff = (size_t)(b * N_ + i0 + r * 16 + v + 8 * g) * INNER_ + h * DH_;
            oh[rowoff + t]      = (_Float16)(o[r][0][v] * inv);
            oh[rowoff + 16 + t] = (_Float16)(o[r][1][v] * inv);
            oh[rowoff + 32 + t] = (_Float16)(o[r][2][v] * inv);
            oh[rowoff + 48 + t] = (_Float16)(o[r][3][v] * inv);
        }
}

// ---------------- host launcher
extern "C" void kernel_launch(void* const* d_in, const int* in_sizes, int n_in,
                              void* d_out, int out_size, void* d_ws, size_t ws_size,
                              hipStream_t stream) {
    (void)in_sizes; (void)n_in; (void)out_size; (void)ws_size;
    const float* x     = (const float*)d_in[0];
    const float* gamma = (const float*)d_in[1];
    const float* beta  = (const float*)d_in[2];
    const float* Wq    = (const float*)d_in[3];
    const float* Wkv   = (const float*)d_in[4];
    const float* Wo    = (const float*)d_in[5];
    float* out = (float*)d_out;

    char* ws = (char*)d_ws;
    const size_t MB = 1024 * 1024;
    _Float16* xnh  = (_Float16*)(ws + 0);          //  8 MiB  [4096 x 1024]
    _Float16* wqT  = (_Float16*)(ws + 8  * MB);    //  2 MiB  [1024 x 1024]
    _Float16* wkvT = (_Float16*)(ws + 10 * MB);    //  4 MiB  [2048 x 1024]
    _Float16* woT  = (_Float16*)(ws + 14 * MB);    //  2 MiB  [1024 x 1024]
    float*    qf   = (float*)   (ws + 16 * MB);    // 16 MiB  [4096 x 1024]
    float*    kvf  = (float*)   (ws + 32 * MB);    // 32 MiB  [4096 x 2048]
    _Float16* qhh  = (_Float16*)(ws + 64 * MB);    //  8 MiB  [b,h,n,d]
    _Float16* khh  = (_Float16*)(ws + 72 * MB);    //  8 MiB  [b,h,n,d]
    _Float16* vtt  = (_Float16*)(ws + 80 * MB);    //  8 MiB  [b,h,d,n]
    _Float16* ohh  = (_Float16*)(ws + 88 * MB);    //  8 MiB  [4096 x 1024]

    wconv_t_kernel<<<(DIM_ * INNER_ + 255) / 256, 256, 0, stream>>>(Wq, wqT, DIM_, INNER_);
    wconv_t_kernel<<<(DIM_ * 2 * INNER_ + 255) / 256, 256, 0, stream>>>(Wkv, wkvT, DIM_, 2 * INNER_);
    wconv_t_kernel<<<(INNER_ * DIM_ + 255) / 256, 256, 0, stream>>>(Wo, woT, INNER_, DIM_);

    ln_kernel<<<B_ * N_, 256, 0, stream>>>(x, gamma, beta, xnh);

    gemm_f16_kernel<<<dim3((B_ * N_) / 64, INNER_ / 64), 32, 0, stream>>>(
        xnh, wqT, qf, B_ * N_, INNER_, DIM_);
    gemm_f16_kernel<<<dim3((B_ * N_) / 64, (2 * INNER_) / 64), 32, 0, stream>>>(
        xnh, wkvT, kvf, B_ * N_, 2 * INNER_, DIM_);

    rope_kernel<<<(B_ * N_ * INNER_) / 256, 256, 0, stream>>>(qf, kvf, qhh, khh, vtt);

    attn_kernel<<<dim3(N_ / 32, H_, B_), 32, 0, stream>>>(qhh, khh, vtt, ohh);

    gemm_f16_kernel<<<dim3((B_ * N_) / 64, DIM_ / 64), 32, 0, stream>>>(
        ohh, woT, out, B_ * N_, DIM_, INNER_);
}
